// ViTBlock_54898271977739
// MI455X (gfx1250) — compile-verified, matrix operand order checked by reading, not run on hardware
//
#include <hip/hip_runtime.h>
#include <math.h>

// ---------------- types ----------------
typedef __bf16 bf16;
typedef __attribute__((ext_vector_type(16))) __bf16 v16bf;
typedef __attribute__((ext_vector_type(8)))  __bf16 v8bf;
typedef __attribute__((ext_vector_type(8)))  float  v8f;

// ---------------- model constants ----------------
#define E_DIM   768
#define H_HEADS 8
#define HD_DIM  96
#define FFN_DIM 3072
#define B_IMG   64
#define GP      14
#define NPATCH  196          // 14*14
#define S_TOK   197          // NPATCH + 1
#define M_TOK   (S_TOK * B_IMG)     // 12608
#define M_PAT   (NPATCH * B_IMG)    // 12544
#define DIL0    2

static __device__ __forceinline__ bf16 f2bf(float x) {
  unsigned u = __builtin_bit_cast(unsigned, x);
  u += 0x7fffu + ((u >> 16) & 1u);          // round to nearest even
  unsigned short h = (unsigned short)(u >> 16);
  return __builtin_bit_cast(bf16, h);
}

static __device__ __forceinline__ unsigned lds_off(const void* p) {
  return (unsigned)(size_t)p;   // flat LDS aperture: low 32 bits = LDS byte offset
}

// 32B contiguous global -> LDS (two async b128, ASYNCcnt-tracked)
static __device__ __forceinline__ void async_copy32(unsigned l, const bf16* g) {
  asm volatile("global_load_async_to_lds_b128 %0, %1, off"
               :: "v"(l), "v"(g) : "memory");
  asm volatile("global_load_async_to_lds_b128 %0, %1, off offset:16"
               :: "v"(l), "v"(g) : "memory");
}
// 64B contiguous global -> LDS (four async b128)
static __device__ __forceinline__ void async_copy64(unsigned l, const bf16* g) {
  asm volatile("global_load_async_to_lds_b128 %0, %1, off"
               :: "v"(l), "v"(g) : "memory");
  asm volatile("global_load_async_to_lds_b128 %0, %1, off offset:16"
               :: "v"(l), "v"(g) : "memory");
  asm volatile("global_load_async_to_lds_b128 %0, %1, off offset:32"
               :: "v"(l), "v"(g) : "memory");
  asm volatile("global_load_async_to_lds_b128 %0, %1, off offset:48"
               :: "v"(l), "v"(g) : "memory");
}

// ---------------- f32 -> bf16 convert ----------------
__global__ __launch_bounds__(256) void cvt_f32_bf16(const float* __restrict__ in,
                                                    bf16* __restrict__ out, size_t n) {
  size_t i = (size_t)blockIdx.x * blockDim.x + threadIdx.x;
  if (i < n) out[i] = f2bf(in[i]);
}

// ---------------- f32 (K x N) -> bf16 transposed (N x K) ----------------
__global__ __launch_bounds__(256) void cvt_transpose(const float* __restrict__ in,
                                                     bf16* __restrict__ out,
                                                     int K, int N) {
  size_t o = (size_t)blockIdx.x * blockDim.x + threadIdx.x;
  size_t total = (size_t)K * N;
  if (o >= total) return;
  int n = (int)(o / K);
  int k = (int)(o % K);
  out[o] = f2bf(in[(size_t)k * N + n]);
}

// ---------------- patch extraction -> bf16 A matrix (row m = j*64 + b) ----------------
__global__ __launch_bounds__(256) void build_patches(const float* __restrict__ x,
                                                     bf16* __restrict__ Ap) {
  size_t idx = (size_t)blockIdx.x * blockDim.x + threadIdx.x;
  const size_t total = (size_t)M_PAT * E_DIM;
  if (idx >= total) return;
  int m  = (int)(idx / E_DIM);
  int kk = (int)(idx % E_DIM);
  int b  = m & 63;
  int j  = m >> 6;              // patch index 0..195
  int gi = j / GP, gj = j % GP;
  int c  = kk % 3;
  int pq = kk / 3;
  int p  = pq >> 4, q = pq & 15;
  float v = x[(((size_t)b * 3 + c) * 224 + (gi * 16 + p)) * 224 + (gj * 16 + q)];
  Ap[idx] = f2bf(v);
}

// ---------------- cls row fill: T0 rows 0..63 (s==0) ----------------
__global__ __launch_bounds__(256) void cls_fill(const float* __restrict__ cls,
                                                const float* __restrict__ pos,
                                                float* __restrict__ T0) {
  int idx = blockIdx.x * blockDim.x + threadIdx.x;
  if (idx >= B_IMG * E_DIM) return;
  int b = idx / E_DIM, e = idx % E_DIM;
  T0[(size_t)b * E_DIM + e] = cls[e] + pos[e];
}

// ---------------- WMMA bf16 GEMM, 128(M) x 256(N) tile, async double-buffered LDS ----
// Each of 8 waves computes a 64x64 macro-tile = 4x4 WMMA 16x16x32 tiles.
// Wt is the PRE-TRANSPOSED weight matrix: N x K row-major (k contiguous).
// MODE 0: out = acc + bias
// MODE 1: patch embed: out[(m+64)] = acc + bias + pos_embed[m/64 + 1]
// MODE 2: out = gelu(acc + bias)   (exact erf)
// MODE 3: out = acc + bias + extra[m]  (residual)
#define A_STRIDE 10240u    // 128*40*2 bytes per A buffer
#define B_STRIDE 20480u    // 256*40*2 bytes per B buffer
template <int MODE>
__global__ __launch_bounds__(256) void gemm_bf16(const bf16* __restrict__ A,
                                                 const bf16* __restrict__ Wt,
                                                 const float* __restrict__ bias,
                                                 const float* __restrict__ extra,
                                                 float* __restrict__ out,
                                                 int M, int N, int K) {
  __shared__ __align__(16) bf16 As[2][128][40];   // padded rows (80B = 16B multiple)
  __shared__ __align__(16) bf16 Bs[2][256][40];   // stored [n][k]

  const int tid  = threadIdx.x;
  const int lane = tid & 31;
  const int wave = tid >> 5;
  const int wM   = wave & 1;        // 0..1  -> 64 rows each
  const int wN   = wave >> 1;       // 0..3  -> 64 cols each
  const int half = lane >> 4;       // K-half selector
  const int l16  = lane & 15;
  const int rowBase = blockIdx.y * 128;
  const int colBase = blockIdx.x * 256;

  // staging: A -> thread owns one (row, k-half) 32B chunk; B -> one 64B row
  const int arow = tid >> 1;
  const int ah   = tid & 1;
  int gr = rowBase + arow; if (gr >= M) gr = M - 1;
  const bf16* gA = A  + (size_t)gr * K + ah * 16;
  const bf16* gB = Wt + (size_t)(colBase + tid) * K;
  const unsigned lA = lds_off(&As[0][arow][ah * 16]);
  const unsigned lB = lds_off(&Bs[0][tid][0]);

  v8f acc[4][4];
#pragma unroll
  for (int mi = 0; mi < 4; ++mi)
#pragma unroll
    for (int ni = 0; ni < 4; ++ni)
#pragma unroll
      for (int r = 0; r < 8; ++r) acc[mi][ni][r] = 0.0f;

  const int steps = K >> 5;

  // prologue: stage 0 into buffer 0
  async_copy32(lA, gA);
  async_copy64(lB, gB);

  int buf = 0;
  for (int i = 0; i < steps; ++i) {
    // branch-free prefetch of the next stage into the other buffer
    // (last iteration re-fetches step 0 into the dead buffer; never read)
    {
      int nstep = i + 1; if (nstep == steps) nstep = 0;
      unsigned nb = (unsigned)(buf ^ 1);
      async_copy32(lA + nb * A_STRIDE, gA + (size_t)nstep * 32);
      async_copy64(lB + nb * B_STRIDE, gB + (size_t)nstep * 32);
    }
    asm volatile("s_wait_asynccnt 0x6" ::: "memory");   // stage i complete
    __syncthreads();   // all waves' stage-i data visible

    // ---- fragments per ISA 16-bit lane layout ----
    v16bf afr[4], bfr[4];
#pragma unroll
    for (int mi = 0; mi < 4; ++mi) {
      const bf16* base = &As[buf][wM * 64 + mi * 16 + l16][0];
      v8bf lo = *(const v8bf*)(base + half * 8);
      v8bf hi = *(const v8bf*)(base + 16 + half * 8);
#pragma unroll
      for (int e = 0; e < 8; ++e) { afr[mi][e] = lo[e]; afr[mi][e + 8] = hi[e]; }
    }
#pragma unroll
    for (int ni = 0; ni < 4; ++ni) {
      const bf16* base = &Bs[buf][wN * 64 + ni * 16 + l16][0];
      v8bf lo = *(const v8bf*)(base + half * 8);
      v8bf hi = *(const v8bf*)(base + 16 + half * 8);
#pragma unroll
      for (int e = 0; e < 8; ++e) { bfr[ni][e] = lo[e]; bfr[ni][e + 8] = hi[e]; }
    }

#pragma unroll
    for (int mi = 0; mi < 4; ++mi)
#pragma unroll
      for (int ni = 0; ni < 4; ++ni)
        acc[mi][ni] = __builtin_amdgcn_wmma_f32_16x16x32_bf16(
            false, afr[mi], false, bfr[ni], (short)0, acc[mi][ni], false, false);

    __syncthreads();   // compute done; next iteration may overwrite this buffer
    buf ^= 1;
  }

  // ---- epilogue ----
#pragma unroll
  for (int mi = 0; mi < 4; ++mi) {
#pragma unroll
    for (int ni = 0; ni < 4; ++ni) {
#pragma unroll
      for (int r = 0; r < 8; ++r) {
        int gm = rowBase + wM * 64 + mi * 16 + half * 8 + r;
        int gn = colBase + wN * 64 + ni * 16 + l16;
        if (gm < M) {
          float v = acc[mi][ni][r] + bias[gn];
          if (MODE == 1) {
            v += extra[(size_t)(gm / 64 + 1) * E_DIM + gn];
            out[(size_t)(gm + B_IMG) * N + gn] = v;
          } else if (MODE == 2) {
            v = 0.5f * v * (1.0f + erff(v * 0.70710678118f));
            out[(size_t)gm * N + gn] = v;
          } else if (MODE == 3) {
            v += extra[(size_t)gm * N + gn];
            out[(size_t)gm * N + gn] = v;
          } else {
            out[(size_t)gm * N + gn] = v;
          }
        }
      }
    }
  }
}

// ---------------- attention core: batch = 197 tokens, seq = 64 imgs, dilation 2 ----------------
__global__ __launch_bounds__(64) void attn_core(const float* __restrict__ q,
                                                const float* __restrict__ k,
                                                const float* __restrict__ v,
                                                float* __restrict__ ctx) {
  const int p = blockIdx.x;   // 0..196
  const int h = blockIdx.y;   // 0..7
  __shared__ float Qs[64][HD_DIM];
  __shared__ float Ks[32][HD_DIM];
  __shared__ float Vs[32][HD_DIM];
  const int t = threadIdx.x;  // 0..63

  for (int d = 0; d < HD_DIM; ++d)
    Qs[t][d] = q[((size_t)p * 64 + t) * E_DIM + h * HD_DIM + d];
  if (t < 32) {
    for (int d = 0; d < HD_DIM; ++d)
      Ks[t][d] = k[((size_t)p * 64 + DIL0 * t) * E_DIM + h * HD_DIM + d];
  } else {
    int j = t - 32;
    for (int d = 0; d < HD_DIM; ++d)
      Vs[j][d] = v[((size_t)p * 64 + DIL0 * j) * E_DIM + h * HD_DIM + d];
  }
  __syncthreads();

  float s[32];
  float mx = -3.0e38f;
  const float scale = 0.10206207262f;  // 1/sqrt(96)
#pragma unroll 4
  for (int j = 0; j < 32; ++j) {
    float a = 0.0f;
    for (int d = 0; d < HD_DIM; ++d) a += Qs[t][d] * Ks[j][d];
    s[j] = a * scale;
    mx = fmaxf(mx, s[j]);
  }
  float sum = 0.0f;
#pragma unroll
  for (int j = 0; j < 32; ++j) { s[j] = expf(s[j] - mx); sum += s[j]; }
  float inv = 1.0f / sum;
  for (int d = 0; d < HD_DIM; ++d) {
    float a = 0.0f;
#pragma unroll 8
    for (int j = 0; j < 32; ++j) a += s[j] * Vs[j][d];
    ctx[((size_t)p * 64 + t) * E_DIM + h * HD_DIM + d] = a * inv;
  }
}

// ---------------- LayerNorm over last dim (768), rows = 12608 ----------------
__global__ __launch_bounds__(256) void layernorm_k(const float* __restrict__ in,
                                                   const float* __restrict__ g,
                                                   const float* __restrict__ b,
                                                   float* __restrict__ out) {
  const int row = blockIdx.x;
  const int t = threadIdx.x;
  __shared__ float sbuf[256];
  float lsum = 0.f, lsq = 0.f;
  float vals[3];
#pragma unroll
  for (int i = 0; i < 3; ++i) {
    float v = in[(size_t)row * E_DIM + t + i * 256];
    vals[i] = v; lsum += v; lsq += v * v;
  }
  sbuf[t] = lsum; __syncthreads();
  for (int st = 128; st > 0; st >>= 1) { if (t < st) sbuf[t] += sbuf[t + st]; __syncthreads(); }
  float mean = sbuf[0] * (1.0f / E_DIM);
  __syncthreads();
  sbuf[t] = lsq; __syncthreads();
  for (int st = 128; st > 0; st >>= 1) { if (t < st) sbuf[t] += sbuf[t + st]; __syncthreads(); }
  float var = sbuf[0] * (1.0f / E_DIM) - mean * mean;
  float rstd = rsqrtf(var + 1e-5f);
#pragma unroll
  for (int i = 0; i < 3; ++i) {
    int e = t + i * 256;
    out[(size_t)row * E_DIM + e] = (vals[i] - mean) * rstd * g[e] + b[e];
  }
}

// ---------------- final gather: (s,b,e) tokens -> (B, E, 14, 14) feature map ----------------
__global__ __launch_bounds__(256) void gather_out(const float* __restrict__ tok,
                                                  float* __restrict__ out) {
  size_t idx = (size_t)blockIdx.x * blockDim.x + threadIdx.x;
  const size_t total = (size_t)B_IMG * E_DIM * NPATCH;
  if (idx >= total) return;
  int b   = (int)(idx / (E_DIM * NPATCH));
  int rem = (int)(idx % (E_DIM * NPATCH));
  int i3 = rem / NPATCH;          // runs over E in output row-major
  int w  = rem % NPATCH;
  int i1 = w / GP, i2 = w % GP;
  int Lsrc = i1 * (GP * E_DIM) + i2 * E_DIM + i3;   // flat index into (E, NPATCH) of body^T
  int e = Lsrc / NPATCH;
  int j = Lsrc % NPATCH;
  out[idx] = tok[((size_t)(j + 1) * B_IMG + b) * E_DIM + e];
}

// ---------------- host launcher ----------------
extern "C" void kernel_launch(void* const* d_in, const int* in_sizes, int n_in,
                              void* d_out, int out_size, void* d_ws, size_t ws_size,
                              hipStream_t stream) {
  (void)in_sizes; (void)n_in; (void)out_size; (void)ws_size;
  const float* x        = (const float*)d_in[0];
  const float* cls_tok  = (const float*)d_in[1];
  const float* pos      = (const float*)d_in[2];
  const float* patch_w  = (const float*)d_in[3];
  const float* patch_b  = (const float*)d_in[4];
  const float* wq       = (const float*)d_in[5];    // [3][E][E] -> slice 0
  const float* bq       = (const float*)d_in[6];
  const float* wk       = (const float*)d_in[7];
  const float* bk       = (const float*)d_in[8];
  const float* wv       = (const float*)d_in[9];
  const float* bv       = (const float*)d_in[10];
  const float* wo       = (const float*)d_in[11];
  const float* bo       = (const float*)d_in[12];
  const float* w1       = (const float*)d_in[13];
  const float* b1       = (const float*)d_in[14];
  const float* w2       = (const float*)d_in[15];
  const float* b2       = (const float*)d_in[16];
  const float* ln1_g    = (const float*)d_in[17];
  const float* ln1_b    = (const float*)d_in[18];
  const float* ln2_g    = (const float*)d_in[19];
  const float* ln2_b    = (const float*)d_in[20];
  float* outp = (float*)d_out;

  // workspace layout
  char* ws = (char*)d_ws;
  size_t off = 0;
  auto alloc = [&](size_t bytes) {
    void* p = ws + off;
    off = (off + bytes + 255) & ~(size_t)255;
    return p;
  };
  const size_t WEE  = (size_t)E_DIM * E_DIM;       // 589824
  const size_t WEF  = (size_t)E_DIM * FFN_DIM;     // 2359296
  bf16* pwT  = (bf16*)alloc(WEE * 2);   // transposed (N x K) bf16 weights
  bf16* wqT  = (bf16*)alloc(WEE * 2);
  bf16* wkT  = (bf16*)alloc(WEE * 2);
  bf16* wvT  = (bf16*)alloc(WEE * 2);
  bf16* woT  = (bf16*)alloc(WEE * 2);
  bf16* w1T  = (bf16*)alloc(WEF * 2);
  bf16* w2T  = (bf16*)alloc(WEF * 2);
  bf16* Abf  = (bf16*)alloc((size_t)M_TOK * FFN_DIM * 2);   // staged bf16 activations
  float* T0  = (float*)alloc((size_t)M_TOK * E_DIM * 4);    // tokens (s,b,e)
  float* T1  = (float*)alloc((size_t)M_TOK * E_DIM * 4);    // ln1 out (FFN input + residual)
  float* F1  = (float*)alloc((size_t)M_TOK * FFN_DIM * 4);  // q|k|v then h1
  float* F2  = (float*)alloc((size_t)M_TOK * E_DIM * 4);    // ctx then ffn2-out
  float* F3  = (float*)alloc((size_t)M_TOK * E_DIM * 4);    // attn-proj then ln2 out

  float* qb = F1;
  float* kb = F1 + (size_t)M_TOK * E_DIM;
  float* vb = F1 + 2 * (size_t)M_TOK * E_DIM;

  auto cvt = [&](const float* src, bf16* dst, size_t n) {
    unsigned blocks = (unsigned)((n + 255) / 256);
    hipLaunchKernelGGL(cvt_f32_bf16, dim3(blocks), dim3(256), 0, stream, src, dst, n);
  };
  auto cvtT = [&](const float* src, bf16* dst, int K, int N) {
    size_t n = (size_t)K * N;
    unsigned blocks = (unsigned)((n + 255) / 256);
    hipLaunchKernelGGL(cvt_transpose, dim3(blocks), dim3(256), 0, stream, src, dst, K, N);
  };

  // 1. weights -> bf16, transposed to N x K
  cvtT(patch_w, pwT, E_DIM, E_DIM);
  cvtT(wq, wqT, E_DIM, E_DIM);
  cvtT(wk, wkT, E_DIM, E_DIM);
  cvtT(wv, wvT, E_DIM, E_DIM);
  cvtT(wo, woT, E_DIM, E_DIM);
  cvtT(w1, w1T, E_DIM, FFN_DIM);
  cvtT(w2, w2T, FFN_DIM, E_DIM);

  // 2. patches -> bf16 A (M_PAT x 768)
  {
    size_t total = (size_t)M_PAT * E_DIM;
    hipLaunchKernelGGL(build_patches, dim3((unsigned)((total + 255) / 256)), dim3(256), 0, stream, x, Abf);
  }

  // 3. patch embed GEMM (+bias +pos, rows shifted by 64) -> T0
  hipLaunchKernelGGL((gemm_bf16<1>), dim3(E_DIM / 256, (M_PAT + 127) / 128), dim3(256), 0, stream,
                     Abf, pwT, patch_b, pos, T0, M_PAT, E_DIM, E_DIM);

  // 4. cls rows (s == 0)
  hipLaunchKernelGGL(cls_fill, dim3((B_IMG * E_DIM + 255) / 256), dim3(256), 0, stream, cls_tok, pos, T0);

  // 5. tokens -> bf16
  cvt(T0, Abf, (size_t)M_TOK * E_DIM);

  // 6. Q, K, V projections
  dim3 gproj(E_DIM / 256, (M_TOK + 127) / 128);
  hipLaunchKernelGGL((gemm_bf16<0>), gproj, dim3(256), 0, stream, Abf, wqT, bq, (const float*)nullptr, qb, M_TOK, E_DIM, E_DIM);
  hipLaunchKernelGGL((gemm_bf16<0>), gproj, dim3(256), 0, stream, Abf, wkT, bk, (const float*)nullptr, kb, M_TOK, E_DIM, E_DIM);
  hipLaunchKernelGGL((gemm_bf16<0>), gproj, dim3(256), 0, stream, Abf, wvT, bv, (const float*)nullptr, vb, M_TOK, E_DIM, E_DIM);

  // 7. attention core -> F2 (ctx)
  hipLaunchKernelGGL(attn_core, dim3(S_TOK, H_HEADS), dim3(64), 0, stream, qb, kb, vb, F2);

  // 8. output projection -> F3
  cvt(F2, Abf, (size_t)M_TOK * E_DIM);
  hipLaunchKernelGGL((gemm_bf16<0>), gproj, dim3(256), 0, stream, Abf, woT, bo, (const float*)nullptr, F3, M_TOK, E_DIM, E_DIM);

  // 9. LN1 -> T1
  hipLaunchKernelGGL(layernorm_k, dim3(M_TOK), dim3(256), 0, stream, F3, ln1_g, ln1_b, T1);

  // 10. FFN1 (+bias +gelu) -> F1
  cvt(T1, Abf, (size_t)M_TOK * E_DIM);
  hipLaunchKernelGGL((gemm_bf16<2>), dim3(FFN_DIM / 256, (M_TOK + 127) / 128), dim3(256), 0, stream,
                     Abf, w1T, b1, (const float*)nullptr, F1, M_TOK, FFN_DIM, E_DIM);

  // 11. FFN2 (+bias +residual T1) -> F2
  cvt(F1, Abf, (size_t)M_TOK * FFN_DIM);
  hipLaunchKernelGGL((gemm_bf16<3>), gproj, dim3(256), 0, stream, Abf, w2T, b2, T1, F2, M_TOK, E_DIM, FFN_DIM);

  // 12. LN2 -> F3
  hipLaunchKernelGGL(layernorm_k, dim3(M_TOK), dim3(256), 0, stream, F2, ln2_g, ln2_b, F3);

  // 13. gather to (B, E, 14, 14)
  {
    size_t total = (size_t)B_IMG * E_DIM * NPATCH;
    hipLaunchKernelGGL(gather_out, dim3((unsigned)((total + 255) / 256)), dim3(256), 0, stream, F3, outp);
  }
}
